// EncoderBlock_21895743275690
// MI455X (gfx1250) — matrix operand order repaired and checked by reading, not compiled, not run on hardware
//
#include <hip/hip_runtime.h>
#include <cstdint>
#include <cstddef>

typedef _Float16 h16;
typedef __attribute__((ext_vector_type(8)))  _Float16 v8h;
typedef __attribute__((ext_vector_type(16))) _Float16 v16h;
typedef __attribute__((ext_vector_type(8)))  float    v8f;

#define BDIM   2
#define SSEQ   2048
#define DMODEL 1024
#define NHEAD  16
#define DKH    64
#define DFFN   4096
#define EPSLN  1e-5f

#define GF_RELU     1
#define GF_OUTF     2
#define GF_H_PLAIN  4
#define GF_H_HEADS  8
#define GF_H_HEADST 16

static __device__ __forceinline__ v16h cat16(v8h lo, v8h hi) {
  return __builtin_shufflevector(lo, hi, 0,1,2,3,4,5,6,7,8,9,10,11,12,13,14,15);
}

// ---------------------------------------------------------------------------
// f32 -> f16 convert
// ---------------------------------------------------------------------------
__global__ void cvt_f32_f16(const float* __restrict__ in, h16* __restrict__ out, int n) {
  int i = blockIdx.x * blockDim.x + threadIdx.x;
  if (i < n) out[i] = (h16)in[i];
}

// ---------------------------------------------------------------------------
// Tiled WMMA GEMM: C[M,N] = A[M,K](f16) @ B[K,N](f16) + bias, epilogue flags.
// 128x128 tile per block (256 threads = 8 waves).  Waves in a 4x2 grid, each
// wave owns a 32x64 slab: 2 A-frags x 4 B-frags -> 8 WMMAs per 32-deep K step
// (12 ds_load_b128 per wave per step -> 1.5 loads/WMMA).
// ---------------------------------------------------------------------------
__global__ __launch_bounds__(256)
void gemm_wmma(const h16* __restrict__ A, const h16* __restrict__ Bm,
               const float* __restrict__ bias,
               float* __restrict__ outF, h16* __restrict__ outH,
               int M, int N, int K, int flags)
{
  __shared__ __align__(16) h16 la[128 * 40];
  __shared__ __align__(16) h16 lb[128 * 40];
  const int t    = threadIdx.x;
  const int lane = t & 31, w = t >> 5;
  const int wr   = w >> 1, wc = w & 1;
  const int hi   = lane >> 4, l16 = lane & 15;
  const int rowBase = blockIdx.y * 128;
  const int colBase = blockIdx.x * 128;

  v8f acc[2][4];
#pragma unroll
  for (int mi = 0; mi < 2; ++mi)
#pragma unroll
    for (int nj = 0; nj < 4; ++nj) acc[mi][nj] = {};

  const int ar = t >> 1, ac = (t & 1) * 16;  // A tile: 128 rows x 32 cols, 2 v8h per thread
  const int bk = t >> 3, bn = (t & 7) * 16;  // B tile: 32 rows x 128 cols, 2 v8h per thread

  for (int k0 = 0; k0 < K; k0 += 32) {
    __syncthreads();
    v8h a0 = *(const v8h*)(A + (size_t)(rowBase + ar) * K + k0 + ac);
    v8h a1 = *(const v8h*)(A + (size_t)(rowBase + ar) * K + k0 + ac + 8);
    *(v8h*)(la + ar * 40 + ac)     = a0;
    *(v8h*)(la + ar * 40 + ac + 8) = a1;
    v8h b0 = *(const v8h*)(Bm + (size_t)(k0 + bk) * N + colBase + bn);
    v8h b1 = *(const v8h*)(Bm + (size_t)(k0 + bk) * N + colBase + bn + 8);
#pragma unroll
    for (int i = 0; i < 8; ++i) {
      lb[(bn + i) * 40 + bk]     = b0[i];   // transpose into [n][k]
      lb[(bn + 8 + i) * 40 + bk] = b1[i];
    }
    if (k0 + 32 < K)
      __builtin_prefetch(A + (size_t)(rowBase + ar) * K + k0 + 32 + ac, 0, 0);
    __syncthreads();

    v16h af[2];
#pragma unroll
    for (int mi = 0; mi < 2; ++mi) {
      const h16* arow = la + (wr*32 + mi*16 + l16) * 40;
      af[mi] = cat16(*(const v8h*)(arow + hi*16), *(const v8h*)(arow + hi*16 + 8));
    }
#pragma unroll
    for (int nj = 0; nj < 4; ++nj) {
      const h16* brow = lb + (wc*64 + nj*16 + l16) * 40;
      v16h bf = cat16(*(const v8h*)(brow + hi*16), *(const v8h*)(brow + hi*16 + 8));
      acc[0][nj] = __builtin_amdgcn_wmma_f32_16x16x32_f16(false, af[0], false, bf, (short)0, acc[0][nj], false, false);
      acc[1][nj] = __builtin_amdgcn_wmma_f32_16x16x32_f16(false, af[1], false, bf, (short)0, acc[1][nj], false, false);
    }
  }

#pragma unroll
  for (int mi = 0; mi < 2; ++mi)
#pragma unroll
    for (int nj = 0; nj < 4; ++nj) {
      int gn = colBase + wc*64 + nj*16 + l16;
      float bs = bias ? bias[gn] : 0.f;
#pragma unroll
      for (int r = 0; r < 8; ++r) {
        int gm = rowBase + wr*32 + mi*16 + hi*8 + r;
        float v = acc[mi][nj][r] + bs;
        if (flags & GF_RELU) v = fmaxf(v, 0.f);
        if (flags & GF_OUTF)    outF[(size_t)gm * N + gn] = v;
        if (flags & GF_H_PLAIN) outH[(size_t)gm * N + gn] = (h16)v;
        if (flags & GF_H_HEADS) {            // [B,H,S,DK]
          int b = gm >> 11, s = gm & (SSEQ - 1);
          int h = gn >> 6,  dk = gn & (DKH - 1);
          outH[(((size_t)(b*NHEAD + h)) * SSEQ + s) * DKH + dk] = (h16)v;
        }
        if (flags & GF_H_HEADST) {           // [B,H,DK,S] (for V^T)
          int b = gm >> 11, s = gm & (SSEQ - 1);
          int h = gn >> 6,  dk = gn & (DKH - 1);
          outH[(((size_t)(b*NHEAD + h)) * DKH + dk) * SSEQ + s] = (h16)v;
        }
      }
    }
}

// ---------------------------------------------------------------------------
// Flash attention: one wave owns 16 query rows; streams 32-key blocks.
// Q,K in [B,H,S,DK] f16; V pre-transposed [B,H,DK,S] f16; ctx -> [B,S,D] f16.
// ---------------------------------------------------------------------------
__global__ __launch_bounds__(256)
void attention_wmma(const h16* __restrict__ Qh, const h16* __restrict__ Kh,
                    const h16* __restrict__ VTh, const int* __restrict__ mask,
                    h16* __restrict__ ctxh)
{
  __shared__ __align__(16) h16 pstrip[8][16 * 32];
  const int t    = threadIdx.x;
  const int lane = t & 31, w = t >> 5;
  const int hi   = lane >> 4, l16 = lane & 15;
  const int head = blockIdx.y;
  const int b    = head / NHEAD, h = head - b * NHEAD;
  const int q0   = blockIdx.x * 128 + w * 16;

  const h16* qbase  = Qh  + (size_t)head * SSEQ * DKH;
  const h16* kbase  = Kh  + (size_t)head * SSEQ * DKH;
  const h16* vtbase = VTh + (size_t)head * DKH * SSEQ;
  const int* mrp    = mask + b * SSEQ;
  h16* strip = pstrip[w];

  const h16* qrow = qbase + (size_t)(q0 + l16) * DKH;
  v16h qa0 = cat16(*(const v8h*)(qrow +      hi*16), *(const v8h*)(qrow +      hi*16 + 8));
  v16h qa1 = cat16(*(const v8h*)(qrow + 32 + hi*16), *(const v8h*)(qrow + 32 + hi*16 + 8));

  v8f o0 = {}, o1 = {}, o2 = {}, o3 = {};
  float mr[8], lr[8];
#pragma unroll
  for (int r = 0; r < 8; ++r) { mr[r] = -1e30f; lr[r] = 0.f; }
  const float scale = 0.125f;  // 1/sqrt(DK)

  for (int k0 = 0; k0 < SSEQ; k0 += 32) {
    v8f s0 = {}, s1 = {};
    {
      const h16* kr = kbase + (size_t)(k0 + l16) * DKH;
      v16h b0 = cat16(*(const v8h*)(kr +      hi*16), *(const v8h*)(kr +      hi*16 + 8));
      v16h b1 = cat16(*(const v8h*)(kr + 32 + hi*16), *(const v8h*)(kr + 32 + hi*16 + 8));
      s0 = __builtin_amdgcn_wmma_f32_16x16x32_f16(false, qa0, false, b0, (short)0, s0, false, false);
      s0 = __builtin_amdgcn_wmma_f32_16x16x32_f16(false, qa1, false, b1, (short)0, s0, false, false);
    }
    {
      const h16* kr = kbase + (size_t)(k0 + 16 + l16) * DKH;
      v16h b0 = cat16(*(const v8h*)(kr +      hi*16), *(const v8h*)(kr +      hi*16 + 8));
      v16h b1 = cat16(*(const v8h*)(kr + 32 + hi*16), *(const v8h*)(kr + 32 + hi*16 + 8));
      s1 = __builtin_amdgcn_wmma_f32_16x16x32_f16(false, qa0, false, b0, (short)0, s1, false, false);
      s1 = __builtin_amdgcn_wmma_f32_16x16x32_f16(false, qa1, false, b1, (short)0, s1, false, false);
    }

    int msk0 = mrp[k0 + l16];
    int msk1 = mrp[k0 + 16 + l16];
    float sm0[8], sm1[8], rmax[8];
#pragma unroll
    for (int r = 0; r < 8; ++r) {
      sm0[r] = msk0 ? s0[r] * scale : -1e20f;
      sm1[r] = msk1 ? s1[r] * scale : -1e20f;
      rmax[r] = fmaxf(sm0[r], sm1[r]);
    }
#pragma unroll
    for (int off = 1; off < 16; off <<= 1)
#pragma unroll
      for (int r = 0; r < 8; ++r)
        rmax[r] = fmaxf(rmax[r], __shfl_xor(rmax[r], off, 16));

    float alpha[8], rsum[8];
#pragma unroll
    for (int r = 0; r < 8; ++r) {
      float nm = fmaxf(mr[r], rmax[r]);
      alpha[r] = __expf(mr[r] - nm);
      mr[r] = nm;
      float p0 = __expf(sm0[r] - nm);
      float p1 = __expf(sm1[r] - nm);
      strip[(hi*8 + r) * 32 + l16]      = (h16)p0;
      strip[(hi*8 + r) * 32 + 16 + l16] = (h16)p1;
      rsum[r] = p0 + p1;
    }
#pragma unroll
    for (int off = 1; off < 16; off <<= 1)
#pragma unroll
      for (int r = 0; r < 8; ++r)
        rsum[r] += __shfl_xor(rsum[r], off, 16);
#pragma unroll
    for (int r = 0; r < 8; ++r) {
      lr[r] = lr[r] * alpha[r] + rsum[r];
      o0[r] *= alpha[r]; o1[r] *= alpha[r]; o2[r] *= alpha[r]; o3[r] *= alpha[r];
    }

    asm volatile("s_wait_dscnt 0" ::: "memory");  // in-wave LDS RAW fence

    const h16* prow = strip + l16 * 32;
    v16h pa = cat16(*(const v8h*)(prow + hi*16), *(const v8h*)(prow + hi*16 + 8));

    {
      const h16* vr = vtbase + (size_t)(l16) * SSEQ + k0;
      v16h vb = cat16(*(const v8h*)(vr + hi*16), *(const v8h*)(vr + hi*16 + 8));
      o0 = __builtin_amdgcn_wmma_f32_16x16x32_f16(false, pa, false, vb, (short)0, o0, false, false);
    }
    {
      const h16* vr = vtbase + (size_t)(16 + l16) * SSEQ + k0;
      v16h vb = cat16(*(const v8h*)(vr + hi*16), *(const v8h*)(vr + hi*16 + 8));
      o1 = __builtin_amdgcn_wmma_f32_16x16x32_f16(false, pa, false, vb, (short)0, o1, false, false);
    }
    {
      const h16* vr = vtbase + (size_t)(32 + l16) * SSEQ + k0;
      v16h vb = cat16(*(const v8h*)(vr + hi*16), *(const v8h*)(vr + hi*16 + 8));
      o2 = __builtin_amdgcn_wmma_f32_16x16x32_f16(false, pa, false, vb, (short)0, o2, false, false);
    }
    {
      const h16* vr = vtbase + (size_t)(48 + l16) * SSEQ + k0;
      v16h vb = cat16(*(const v8h*)(vr + hi*16), *(const v8h*)(vr + hi*16 + 8));
      o3 = __builtin_amdgcn_wmma_f32_16x16x32_f16(false, pa, false, vb, (short)0, o3, false, false);
    }
  }

#pragma unroll
  for (int r = 0; r < 8; ++r) {
    float inv = 1.f / lr[r];
    int s = q0 + hi*8 + r;
    size_t base = ((size_t)(b * SSEQ + s)) * DMODEL + h * DKH + l16;
    ctxh[base +  0] = (h16)(o0[r] * inv);
    ctxh[base + 16] = (h16)(o1[r] * inv);
    ctxh[base + 32] = (h16)(o2[r] * inv);
    ctxh[base + 48] = (h16)(o3[r] * inv);
  }
}

// ---------------------------------------------------------------------------
// out = LayerNorm(x + y) * g + be ; optional f16 copy for next GEMM.
// One block (256 thr) per row of D=1024.
// ---------------------------------------------------------------------------
__global__ __launch_bounds__(256)
void add_ln(const float* __restrict__ x, const float* __restrict__ y,
            const float* __restrict__ g, const float* __restrict__ be,
            float* __restrict__ outF, h16* __restrict__ outH)
{
  __shared__ float red[256];
  const int row = blockIdx.x;
  const int t = threadIdx.x;
  const float* xr = x + (size_t)row * DMODEL;
  const float* yr = y + (size_t)row * DMODEL;
  float v[4]; float s = 0.f, s2 = 0.f;
#pragma unroll
  for (int i = 0; i < 4; ++i) {
    int c = t + i * 256;
    float val = xr[c] + yr[c];
    v[i] = val; s += val; s2 += val * val;
  }
  red[t] = s; __syncthreads();
  for (int off = 128; off > 0; off >>= 1) { if (t < off) red[t] += red[t + off]; __syncthreads(); }
  float mean = red[0] * (1.f / DMODEL);
  __syncthreads();
  red[t] = s2; __syncthreads();
  for (int off = 128; off > 0; off >>= 1) { if (t < off) red[t] += red[t + off]; __syncthreads(); }
  float var  = red[0] * (1.f / DMODEL) - mean * mean;
  float rstd = rsqrtf(var + EPSLN);
#pragma unroll
  for (int i = 0; i < 4; ++i) {
    int c = t + i * 256;
    float o = (v[i] - mean) * rstd * g[c] + be[c];
    outF[(size_t)row * DMODEL + c] = o;
    if (outH) outH[(size_t)row * DMODEL + c] = (h16)o;
  }
}

// ---------------------------------------------------------------------------
extern "C" void kernel_launch(void* const* d_in, const int* in_sizes, int n_in,
                              void* d_out, int out_size, void* d_ws, size_t ws_size,
                              hipStream_t stream)
{
  (void)in_sizes; (void)n_in; (void)out_size; (void)ws_size;
  const float* x    = (const float*)d_in[0];
  const int*   mask = (const int*)  d_in[1];
  const float* Wq   = (const float*)d_in[2];
  const float* bq   = (const float*)d_in[3];
  const float* Wk   = (const float*)d_in[4];
  const float* bk   = (const float*)d_in[5];
  const float* Wv   = (const float*)d_in[6];
  const float* bv   = (const float*)d_in[7];
  const float* Wo   = (const float*)d_in[8];
  const float* bo   = (const float*)d_in[9];
  const float* W1   = (const float*)d_in[10];
  const float* b1   = (const float*)d_in[11];
  const float* W2   = (const float*)d_in[12];
  const float* b2   = (const float*)d_in[13];
  const float* g1   = (const float*)d_in[14];
  const float* be1  = (const float*)d_in[15];
  const float* g2   = (const float*)d_in[16];
  const float* be2  = (const float*)d_in[17];

  const int M  = BDIM * SSEQ;            // 4096
  const size_t nXD = (size_t)M * DMODEL; // 4,194,304
  const size_t nDD = (size_t)DMODEL * DMODEL;
  const size_t nDF = (size_t)DMODEL * DFFN;
  const size_t nXF = (size_t)M * DFFN;

  char* p = (char*)d_ws;
  auto carve = [&](size_t bytes) -> void* {
    void* r = (void*)p;
    p += (bytes + 255) & ~(size_t)255;
    return r;
  };
  h16* xh    = (h16*)carve(nXD * 2);
  h16* Wqh   = (h16*)carve(nDD * 2);
  h16* Wkh   = (h16*)carve(nDD * 2);
  h16* Wvh   = (h16*)carve(nDD * 2);
  h16* Woh   = (h16*)carve(nDD * 2);
  h16* W1h   = (h16*)carve(nDF * 2);
  h16* W2h   = (h16*)carve(nDF * 2);
  h16* qh    = (h16*)carve(nXD * 2);
  h16* kh    = (h16*)carve(nXD * 2);
  h16* vth   = (h16*)carve(nXD * 2);
  h16* ctxh  = (h16*)carve(nXD * 2);
  h16* out1h = (h16*)carve(nXD * 2);
  h16* ffh   = (h16*)carve(nXF * 2);
  float* attn_out = (float*)carve(nXD * 4);
  float* out1     = (float*)carve(nXD * 4);
  float* ff2      = (float*)carve(nXD * 4);

  auto cvt = [&](const float* src, h16* dst, size_t n) {
    cvt_f32_f16<<<dim3((unsigned)((n + 255) / 256)), dim3(256), 0, stream>>>(src, dst, (int)n);
  };
  cvt(x,  xh,  nXD);
  cvt(Wq, Wqh, nDD); cvt(Wk, Wkh, nDD); cvt(Wv, Wvh, nDD); cvt(Wo, Woh, nDD);
  cvt(W1, W1h, nDF); cvt(W2, W2h, nDF);

  dim3 blk(256);
  dim3 gDD(DMODEL / 128, M / 128);  // (8, 32)  N=1024 GEMMs
  dim3 gDF(DFFN / 128,  M / 128);   // (32, 32) N=4096 GEMM

  gemm_wmma<<<gDD, blk, 0, stream>>>(xh, Wqh, bq, nullptr, qh,  M, DMODEL, DMODEL, GF_H_HEADS);
  gemm_wmma<<<gDD, blk, 0, stream>>>(xh, Wkh, bk, nullptr, kh,  M, DMODEL, DMODEL, GF_H_HEADS);
  gemm_wmma<<<gDD, blk, 0, stream>>>(xh, Wvh, bv, nullptr, vth, M, DMODEL, DMODEL, GF_H_HEADST);

  dim3 gAtt(SSEQ / 128, BDIM * NHEAD);
  attention_wmma<<<gAtt, blk, 0, stream>>>(qh, kh, vth, mask, ctxh);

  gemm_wmma<<<gDD, blk, 0, stream>>>(ctxh, Woh, bo, attn_out, nullptr, M, DMODEL, DMODEL, GF_OUTF);
  add_ln<<<dim3(M), blk, 0, stream>>>(x, attn_out, g1, be1, out1, out1h);

  gemm_wmma<<<gDF, blk, 0, stream>>>(out1h, W1h, b1, nullptr, ffh, M, DFFN, DMODEL, GF_RELU | GF_H_PLAIN);
  gemm_wmma<<<gDD, blk, 0, stream>>>(ffh, W2h, b2, ff2, nullptr, M, DMODEL, DFFN, GF_OUTF);
  add_ln<<<dim3(M), blk, 0, stream>>>(out1, ff2, g2, be2, (float*)d_out, nullptr);
}
